// AttentionModel_8778913153064
// MI455X (gfx1250) — matrix operand order repaired and checked by reading, not compiled, not run on hardware
//
#include <hip/hip_runtime.h>
#include <stdint.h>

#define B_    256
#define H_    1024
#define F_    512
#define P_    66
#define PPAD  80
#define TIN   10
#define TOUT  25
#define G4H   4096

typedef __attribute__((ext_vector_type(16))) __bf16        bf16x16;
typedef __attribute__((ext_vector_type(8)))  float         f32x8;
typedef __attribute__((ext_vector_type(4)))  unsigned int  u32x4;

union Frag16 { bf16x16 v; u32x4 u[2]; };

__device__ __forceinline__ unsigned short f2bf(float f) {
    union { float f; unsigned int u; } x; x.f = f;
    unsigned int r = x.u + 0x7FFFu + ((x.u >> 16) & 1u);   // round-to-nearest-even
    return (unsigned short)(r >> 16);
}
__device__ __forceinline__ float sigmoidf_(float v) { return 1.0f / (1.0f + __expf(-v)); }

// ---------------- elementwise / prep kernels ----------------

__global__ void fill_f32_kernel(float* p, int n) {
    int gid = blockIdx.x * blockDim.x + threadIdx.x;
    if (gid < n) p[gid] = 0.0f;
}
__global__ void fill_u16_kernel(unsigned short* p, int n) {
    int gid = blockIdx.x * blockDim.x + threadIdx.x;
    if (gid < n) p[gid] = 0;
}
__global__ void convert_bf16_kernel(const float* __restrict__ in, unsigned short* __restrict__ out, int n) {
    int gid = blockIdx.x * blockDim.x + threadIdx.x;
    if (gid < n) out[gid] = f2bf(in[gid]);
}
// tp_w (66,1024) -> padded bf16 (80,1024), rows >=66 zeroed
__global__ void convert_tp_kernel(const float* __restrict__ tpw, unsigned short* __restrict__ out) {
    int gid = blockIdx.x * blockDim.x + threadIdx.x;   // PPAD*H_
    int p = gid >> 10, h = gid & (H_ - 1);
    out[gid] = (p < P_) ? f2bf(tpw[p * H_ + h]) : (unsigned short)0;
}
// ToFeature: x (B,TIN,P) @ tf_w^T + tf_b  -> time-major bf16 (TIN,B,F)
__global__ void to_feature_kernel(const float* __restrict__ x, const float* __restrict__ w,
                                  const float* __restrict__ bias, unsigned short* __restrict__ out) {
    int gid = blockIdx.x * blockDim.x + threadIdx.x;   // TIN*B_*F_
    int f  = gid % F_;
    int tb = gid / F_;
    int t  = tb / B_;
    int b  = tb % B_;
    const float* xr = x + ((size_t)b * TIN + t) * P_;
    const float* wr = w + (size_t)f * P_;
    float s = bias[f];
    for (int k = 0; k < P_; ++k) s += xr[k] * wr[k];
    out[gid] = f2bf(s);
}

// ---------------- WMMA GEMM: out(256,N) = A0(256,K0)@W0(N,K0)^T + A1(256,K1)@W1(N,K1)^T ----------------
// bf16 inputs, f32 accumulate. Each wave: one 16x64 output strip (4 accumulators reuse one A frag).

__device__ __forceinline__ void gemm_seg(const unsigned short* __restrict__ A,
                                         const unsigned short* __restrict__ W, int K,
                                         int row, int colBase, int laneLo, int laneHi,
                                         f32x8 acc[4]) {
    const unsigned short* arow = A + (size_t)row * K;
    const unsigned short* w0 = W + (size_t)(colBase +  0 + laneLo) * K;
    const unsigned short* w1 = W + (size_t)(colBase + 16 + laneLo) * K;
    const unsigned short* w2 = W + (size_t)(colBase + 32 + laneLo) * K;
    const unsigned short* w3 = W + (size_t)(colBase + 48 + laneLo) * K;
    for (int k = 0; k < K; k += 32) {
        // A frag (16x32): lanes 0-15 hold K=0..15, lanes 16-31 hold K=16..31 (contiguous, matches B)
        Frag16 a;
        int koA = k + laneHi * 16;
        a.u[0] = *(const u32x4*)(arow + koA);
        a.u[1] = *(const u32x4*)(arow + koA + 8);
        // B frags (32x16): lanes 0-15 hold K=0..15, lanes 16-31 hold K=16..31 (contiguous)
        int koB = k + laneHi * 16;
        Frag16 b0, b1, b2, b3;
        b0.u[0] = *(const u32x4*)(w0 + koB); b0.u[1] = *(const u32x4*)(w0 + koB + 8);
        b1.u[0] = *(const u32x4*)(w1 + koB); b1.u[1] = *(const u32x4*)(w1 + koB + 8);
        b2.u[0] = *(const u32x4*)(w2 + koB); b2.u[1] = *(const u32x4*)(w2 + koB + 8);
        b3.u[0] = *(const u32x4*)(w3 + koB); b3.u[1] = *(const u32x4*)(w3 + koB + 8);
        acc[0] = __builtin_amdgcn_wmma_f32_16x16x32_bf16(false, a.v, false, b0.v, (short)0, acc[0], false, false);
        acc[1] = __builtin_amdgcn_wmma_f32_16x16x32_bf16(false, a.v, false, b1.v, (short)0, acc[1], false, false);
        acc[2] = __builtin_amdgcn_wmma_f32_16x16x32_bf16(false, a.v, false, b2.v, (short)0, acc[2], false, false);
        acc[3] = __builtin_amdgcn_wmma_f32_16x16x32_bf16(false, a.v, false, b3.v, (short)0, acc[3], false, false);
    }
}

__global__ void wmma_gemm2_kernel(const unsigned short* __restrict__ A0,
                                  const unsigned short* __restrict__ W0, int K0,
                                  const unsigned short* __restrict__ A1,
                                  const unsigned short* __restrict__ W1, int K1,
                                  float* __restrict__ out, int N) {
    int wave = (blockIdx.x * blockDim.x + threadIdx.x) >> 5;
    int lane = threadIdx.x & 31;
    int laneLo = lane & 15, laneHi = lane >> 4;
    int nStrips = N >> 6;
    int tm = wave / nStrips;
    int tn = wave % nStrips;
    int row = tm * 16 + laneLo;
    int colBase = tn * 64;
    f32x8 acc[4] = {};
    gemm_seg(A0, W0, K0, row, colBase, laneLo, laneHi, acc);
    if (K1 > 0) gemm_seg(A1, W1, K1, row, colBase, laneLo, laneHi, acc);
    // C layout: VGPR r -> M = tm*16 + laneHi*8 + r ; N = colBase + j*16 + laneLo
    #pragma unroll
    for (int j = 0; j < 4; ++j) {
        float* orow = out + (size_t)(tm * 16 + laneHi * 8) * N + colBase + j * 16 + laneLo;
        #pragma unroll
        for (int r = 0; r < 8; ++r) orow[(size_t)r * N] = acc[j][r];
    }
}

// ---------------- LSTM elementwise: gates(B,4H)+biases -> h,c ----------------

__global__ void lstm_elem_kernel(const float* __restrict__ gates,
                                 const float* __restrict__ bih, const float* __restrict__ bhh,
                                 float* __restrict__ c, float* __restrict__ h_out,
                                 unsigned short* __restrict__ hbf_out) {
    int gid = blockIdx.x * blockDim.x + threadIdx.x;   // B_*H_
    int b = gid >> 10, j = gid & (H_ - 1);
    const float* g = gates + (size_t)b * G4H;
    float gi = g[j]          + bih[j]          + bhh[j];
    float gf = g[H_ + j]     + bih[H_ + j]     + bhh[H_ + j];
    float gg = g[2 * H_ + j] + bih[2 * H_ + j] + bhh[2 * H_ + j];
    float go = g[3 * H_ + j] + bih[3 * H_ + j] + bhh[3 * H_ + j];
    float cn = sigmoidf_(gf) * c[gid] + sigmoidf_(gi) * tanhf(gg);
    c[gid] = cn;
    float hn = sigmoidf_(go) * tanhf(cn);
    h_out[gid]  = hn;
    hbf_out[gid] = f2bf(hn);
}

// ---------------- attention ----------------

__device__ __forceinline__ const float* cat_row(int t, int b, const float* all_h,
                                                const float* all_hz, const float* h_cur) {
    if (t < 10)  return all_h + ((size_t)t * B_ + b) * H_;
    if (t == 10) return h_cur + (size_t)b * H_;
    return all_hz + ((size_t)(t - 11) * B_ + b) * H_;
}

__global__ void attn_scores_kernel(const float* __restrict__ all_h, const float* __restrict__ all_hz,
                                   const float* __restrict__ h_cur, const float* __restrict__ c,
                                   float* __restrict__ scores) {
    __shared__ float red[256];
    int t = blockIdx.x / B_, b = blockIdx.x % B_;
    const float* hv = cat_row(t, b, all_h, all_hz, h_cur);
    const float* cv = c + (size_t)b * H_;
    float s = 0.f;
    for (int i = threadIdx.x; i < H_; i += 256) s += cv[i] * hv[i];
    red[threadIdx.x] = s; __syncthreads();
    for (int st = 128; st > 0; st >>= 1) {
        if ((int)threadIdx.x < st) red[threadIdx.x] += red[threadIdx.x + st];
        __syncthreads();
    }
    if (threadIdx.x == 0) scores[t * B_ + b] = red[0];
}

__global__ void attn_softmax_kernel(const float* __restrict__ scores, float* __restrict__ attw) {
    int b = threadIdx.x;                               // one block of 256
    float mx = -1e30f;
    for (int t = 0; t < 21; ++t) mx = fmaxf(mx, scores[t * B_ + b]);
    float e[21], sum = 0.f;
    for (int t = 0; t < 21; ++t) { e[t] = __expf(scores[t * B_ + b] - mx); sum += e[t]; }
    float inv = 1.f / sum;
    for (int t = 0; t < 21; ++t) attw[t * B_ + b] = e[t] * inv;
}

__global__ void attn_context_kernel(const float* __restrict__ attw, const float* __restrict__ all_h,
                                    const float* __restrict__ all_hz, const float* __restrict__ h_cur,
                                    unsigned short* __restrict__ atth_bf) {
    int gid = blockIdx.x * blockDim.x + threadIdx.x;   // B_*H_
    int b = gid >> 10, h = gid & (H_ - 1);
    float s = 0.f;
    #pragma unroll
    for (int t = 0; t < 21; ++t) s += attw[t * B_ + b] * cat_row(t, b, all_h, all_hz, h_cur)[h];
    atth_bf[gid] = f2bf(s);
}

__global__ void bias_convert_kernel(const float* __restrict__ in, const float* __restrict__ bias,
                                    unsigned short* __restrict__ out) {
    int gid = blockIdx.x * blockDim.x + threadIdx.x;   // B_*F_
    out[gid] = f2bf(in[gid] + bias[gid & (F_ - 1)]);
}

// ---------------- ToPose WMMA: (25*256,1024)bf16 @ (80,1024)bf16^T + tp_b + resid ----------------

__global__ void to_pose_kernel(const unsigned short* __restrict__ A,
                               const unsigned short* __restrict__ W,
                               const float* __restrict__ tp_b, const float* __restrict__ resid,
                               float* __restrict__ out) {
    const int K = H_, NT = PPAD / 16;                  // 5 N-tiles
    int wave = (blockIdx.x * blockDim.x + threadIdx.x) >> 5;
    int lane = threadIdx.x & 31;
    int laneLo = lane & 15, laneHi = lane >> 4;
    int tm = wave / NT, tn = wave % NT;
    int row = tm * 16 + laneLo;
    f32x8 acc = {};
    const unsigned short* arow = A + (size_t)row * K;
    const unsigned short* wrow = W + (size_t)(tn * 16 + laneLo) * K;
    for (int k = 0; k < K; k += 32) {
        Frag16 a, b;
        int koA = k + laneHi * 16;
        a.u[0] = *(const u32x4*)(arow + koA);
        a.u[1] = *(const u32x4*)(arow + koA + 8);
        int koB = k + laneHi * 16;
        b.u[0] = *(const u32x4*)(wrow + koB);
        b.u[1] = *(const u32x4*)(wrow + koB + 8);
        acc = __builtin_amdgcn_wmma_f32_16x16x32_bf16(false, a.v, false, b.v, (short)0, acc, false, false);
    }
    int n = tn * 16 + laneLo;
    if (n < P_) {
        float bias = tp_b[n];
        #pragma unroll
        for (int r = 0; r < 8; ++r) {
            int m = tm * 16 + laneHi * 8 + r;          // m = t*256 + b
            int t = m >> 8, bb = m & 255;
            size_t o = ((size_t)bb * TOUT + t) * P_ + n;
            out[o] = acc[r] + bias + resid[o];
        }
    }
}

// ---------------- host orchestration ----------------

extern "C" void kernel_launch(void* const* d_in, const int* in_sizes, int n_in,
                              void* d_out, int out_size, void* d_ws, size_t ws_size,
                              hipStream_t stream) {
    const float* x        = (const float*)d_in[0];
    const float* z        = (const float*)d_in[1];
    const float* fresid   = (const float*)d_in[2];
    const float* tf_w     = (const float*)d_in[3];
    const float* tf_b     = (const float*)d_in[4];
    const float* enc_wih  = (const float*)d_in[5];
    const float* enc_whh  = (const float*)d_in[6];
    const float* enc_bih  = (const float*)d_in[7];
    const float* enc_bhh  = (const float*)d_in[8];
    const float* encp_wih = (const float*)d_in[9];
    const float* encp_whh = (const float*)d_in[10];
    const float* encp_bih = (const float*)d_in[11];
    const float* encp_bhh = (const float*)d_in[12];
    const float* dec_wih  = (const float*)d_in[13];
    const float* dec_whh  = (const float*)d_in[14];
    const float* dec_bih  = (const float*)d_in[15];
    const float* dec_bhh  = (const float*)d_in[16];
    const float* lin_w    = (const float*)d_in[17];
    const float* lin_b    = (const float*)d_in[18];
    const float* tp_w     = (const float*)d_in[19];
    const float* tp_b     = (const float*)d_in[20];
    float* out = (float*)d_out;

    size_t off = 0;
    auto alloc = [&](size_t bytes) -> void* {
        off = (off + 255) & ~(size_t)255;
        void* p = (char*)d_ws + off;
        off += bytes;
        return p;
    };
    const size_t BH = (size_t)B_ * H_;
    unsigned short* enc_wih_bf  = (unsigned short*)alloc((size_t)G4H * F_ * 2);
    unsigned short* enc_whh_bf  = (unsigned short*)alloc((size_t)G4H * H_ * 2);
    unsigned short* encp_wih_bf = (unsigned short*)alloc((size_t)G4H * F_ * 2);
    unsigned short* encp_whh_bf = (unsigned short*)alloc((size_t)G4H * H_ * 2);
    unsigned short* dec_wih_bf  = (unsigned short*)alloc((size_t)G4H * F_ * 2);
    unsigned short* dec_whh_bf  = (unsigned short*)alloc((size_t)G4H * H_ * 2);
    unsigned short* lin_w_bf    = (unsigned short*)alloc((size_t)F_ * H_ * 2);
    unsigned short* tp_w_bf     = (unsigned short*)alloc((size_t)PPAD * H_ * 2);
    unsigned short* xf_bf       = (unsigned short*)alloc((size_t)TIN * B_ * F_ * 2);
    unsigned short* zf_bf       = (unsigned short*)alloc((size_t)TIN * B_ * F_ * 2);
    float*          all_h       = (float*)alloc(TIN * BH * 4);
    float*          all_hz      = (float*)alloc(TIN * BH * 4);
    float*          gates       = (float*)alloc((size_t)B_ * G4H * 4);
    float*          enc_c       = (float*)alloc(BH * 4);        // reused as decoder c
    float*          encp_c      = (float*)alloc(BH * 4);
    unsigned short* enc_hbf     = (unsigned short*)alloc(BH * 2);
    unsigned short* encp_hbf    = (unsigned short*)alloc(BH * 2);
    float*          dec_h       = (float*)alloc(BH * 4);
    unsigned short* dec_hs_bf   = (unsigned short*)alloc(TOUT * BH * 2);
    float*          scores      = (float*)alloc(21 * B_ * 4);
    float*          attw        = (float*)alloc(21 * B_ * 4);
    unsigned short* atth_bf     = (unsigned short*)alloc(BH * 2);
    float*          inp_f32     = (float*)alloc((size_t)B_ * F_ * 4);
    unsigned short* inp_bf      = (unsigned short*)alloc((size_t)B_ * F_ * 2);

    auto cvt = [&](const float* src, unsigned short* dst, int n) {
        convert_bf16_kernel<<<(n + 255) / 256, 256, 0, stream>>>(src, dst, n);
    };
    // --- prep ---
    cvt(enc_wih,  enc_wih_bf,  G4H * F_);
    cvt(enc_whh,  enc_whh_bf,  G4H * H_);
    cvt(encp_wih, encp_wih_bf, G4H * F_);
    cvt(encp_whh, encp_whh_bf, G4H * H_);
    cvt(dec_wih,  dec_wih_bf,  G4H * F_);
    cvt(dec_whh,  dec_whh_bf,  G4H * H_);
    cvt(lin_w,    lin_w_bf,    F_ * H_);
    convert_tp_kernel<<<(PPAD * H_) / 256, 256, 0, stream>>>(tp_w, tp_w_bf);
    to_feature_kernel<<<(TIN * B_ * F_) / 256, 256, 0, stream>>>(x, tf_w, tf_b, xf_bf);
    to_feature_kernel<<<(TIN * B_ * F_) / 256, 256, 0, stream>>>(z, tf_w, tf_b, zf_bf);
    fill_f32_kernel<<<(int)(BH / 256), 256, 0, stream>>>(enc_c,  (int)BH);
    fill_f32_kernel<<<(int)(BH / 256), 256, 0, stream>>>(encp_c, (int)BH);
    fill_u16_kernel<<<(int)(BH / 256), 256, 0, stream>>>(enc_hbf,  (int)BH);
    fill_u16_kernel<<<(int)(BH / 256), 256, 0, stream>>>(encp_hbf, (int)BH);

    const int gemmBlocks4096 = (B_ / 16) * (G4H / 64) * 32 / 256;   // 128
    const int gemmBlocks512  = (B_ / 16) * (F_  / 64) * 32 / 256;   // 16
    const int elemBlocks     = (int)(BH / 256);                     // 1024

    // --- encoder LSTMs ---
    for (int t = 0; t < TIN; ++t) {
        wmma_gemm2_kernel<<<gemmBlocks4096, 256, 0, stream>>>(
            xf_bf + (size_t)t * B_ * F_, enc_wih_bf, F_, enc_hbf, enc_whh_bf, H_, gates, G4H);
        lstm_elem_kernel<<<elemBlocks, 256, 0, stream>>>(
            gates, enc_bih, enc_bhh, enc_c, all_h + (size_t)t * BH, enc_hbf);
    }
    for (int t = 0; t < TIN; ++t) {
        wmma_gemm2_kernel<<<gemmBlocks4096, 256, 0, stream>>>(
            zf_bf + (size_t)t * B_ * F_, encp_wih_bf, F_, encp_hbf, encp_whh_bf, H_, gates, G4H);
        lstm_elem_kernel<<<elemBlocks, 256, 0, stream>>>(
            gates, encp_bih, encp_bhh, encp_c, all_hz + (size_t)t * BH, encp_hbf);
    }

    // --- decoder (c continues from enc_c; h from all_h[9]/enc_hbf) ---
    for (int t = 0; t < TOUT; ++t) {
        const float*          h_prev    = (t == 0) ? all_h + 9 * BH : dec_h;
        const unsigned short* h_prev_bf = (t == 0) ? enc_hbf : dec_hs_bf + (size_t)(t - 1) * BH;
        attn_scores_kernel<<<21 * B_, 256, 0, stream>>>(all_h, all_hz, h_prev, enc_c, scores);
        attn_softmax_kernel<<<1, 256, 0, stream>>>(scores, attw);
        attn_context_kernel<<<elemBlocks, 256, 0, stream>>>(attw, all_h, all_hz, h_prev, atth_bf);
        wmma_gemm2_kernel<<<gemmBlocks512, 256, 0, stream>>>(
            h_prev_bf, lin_w_bf, H_, (const unsigned short*)nullptr, (const unsigned short*)nullptr, 0,
            inp_f32, F_);
        bias_convert_kernel<<<(B_ * F_) / 256, 256, 0, stream>>>(inp_f32, lin_b, inp_bf);
        wmma_gemm2_kernel<<<gemmBlocks4096, 256, 0, stream>>>(
            inp_bf, dec_wih_bf, F_, atth_bf, dec_whh_bf, H_, gates, G4H);
        lstm_elem_kernel<<<elemBlocks, 256, 0, stream>>>(
            gates, dec_bih, dec_bhh, enc_c, dec_h, dec_hs_bf + (size_t)t * BH);
    }

    // --- ToPose + residual ---
    const int poseBlocks = ((TOUT * B_) / 16) * (PPAD / 16) * 32 / 256;  // 250
    to_pose_kernel<<<poseBlocks, 256, 0, stream>>>(dec_hs_bf, tp_w_bf, tp_b, fresid, out);
}
